// NetAll_20151986553556
// MI455X (gfx1250) — matrix operand order repaired and checked by reading, not compiled, not run on hardware
//
#include <hip/hip_runtime.h>
#include <stdint.h>

typedef __attribute__((ext_vector_type(16))) _Float16 v16h;
typedef __attribute__((ext_vector_type(8)))  float    v8f;
typedef _Float16 half_t;

#define NNODES 50000
#define NEDGES 800000

enum { ACT_NONE = 0, ACT_SIG = 1, ACT_RELU = 2 };

__device__ __forceinline__ float sigmoidf_(float x) { return 1.f / (1.f + __expf(-x)); }

// ---------------- conversion kernels ----------------
__global__ void k_cvt_slice(const float* __restrict__ xall, int ldx, int c0, int width,
                            half_t* __restrict__ out, int kpad, int n) {
  long idx = (long)blockIdx.x * blockDim.x + threadIdx.x;
  if (idx >= (long)n * kpad) return;
  int r = (int)(idx / kpad), k = (int)(idx - (long)r * kpad);
  float v = (k < width) ? xall[(size_t)r * ldx + c0 + k] : 0.f;
  out[idx] = (half_t)v;
}

__global__ void k_cvt_w(const float* __restrict__ w, int nout, int kin,
                        half_t* __restrict__ out, int noutpad, int kpad) {
  int idx = blockIdx.x * blockDim.x + threadIdx.x;
  if (idx >= noutpad * kpad) return;
  int r = idx / kpad, k = idx - r * kpad;
  float v = (r < nout && k < kin) ? w[r * kin + k] : 0.f;
  out[idx] = (half_t)v;
}

// ---------------- WMMA GEMM: C = act(A(MxK) * W(NoutxK)^T + bias) ----------------
// A row-major f16 with row stride Kpad; W row-major f16 (Nout x Kpad).
// A-frag (16x32 f16): lane m = lane&15; halves 0..7 = K[koff..koff+7],
//                     halves 8..15 = K[koff+16..koff+23], koff = (lane>=16)*8.
// B-frag (32x16 f16): lane n = lane&15; halves 0..7 = W[n][koff..koff+7],
//                     halves 8..15 = W[n][koff+16..koff+23], koff = (lane>=16)*8.
// C-frag (16x16 f32): vgpr r -> row m0 + (lane>=16)*8 + r, col n0 + (lane&15).
template <int ACT, typename OutT>
__global__ __launch_bounds__(256) void k_gemm_wmma(
    const half_t* __restrict__ A, const half_t* __restrict__ W,
    const float* __restrict__ bias, OutT* __restrict__ C,
    int ldc, int col0, int M, int Kpad, int NoutStore) {
  const int wave = threadIdx.x >> 5;
  const int lane = threadIdx.x & 31;
  const int mt = blockIdx.x * 8 + wave;
  if (mt * 16 >= M) return;                       // wave-uniform exit (EXEC stays all-1)
  const int m0 = mt * 16;
  const int hsel = lane >> 4;
  const int mrow = m0 + (lane & 15);
  const int n = lane & 15;
  const int ncol = blockIdx.y * 16 + n;
  const half_t* __restrict__ arow = A + (size_t)mrow * Kpad + hsel * 8;
  const half_t* __restrict__ wrow = W + (size_t)ncol * Kpad + hsel * 8;
  v8f acc = {};
  for (int k0 = 0; k0 < Kpad; k0 += 32) {
    union { v16h v; uint4 u[2]; } ua, ub;
    ua.u[0] = *(const uint4*)(arow + k0);
    ua.u[1] = *(const uint4*)(arow + k0 + 16);
    ub.u[0] = *(const uint4*)(wrow + k0);
    ub.u[1] = *(const uint4*)(wrow + k0 + 16);
    if (k0 + 32 < Kpad) __builtin_prefetch((const void*)(arow + k0 + 32), 0, 1);
    acc = __builtin_amdgcn_wmma_f32_16x16x32_f16(false, ua.v, false, ub.v,
                                                 (short)0, acc, false, false);
  }
  if (ncol < NoutStore) {
    const float bv = bias ? bias[ncol] : 0.f;
    const int mbase = m0 + hsel * 8;
#pragma unroll
    for (int r = 0; r < 8; ++r) {
      float v = acc[r] + bv;
      if (ACT == ACT_SIG) v = sigmoidf_(v);
      if (ACT == ACT_RELU) v = fmaxf(v, 0.f);
      C[(size_t)(mbase + r) * ldc + col0 + ncol] = (OutT)v;
    }
  }
}

// ---------------- GAT softmax / aggregation ----------------
__device__ __forceinline__ void atomicMaxFloat(float* addr, float val) {
  if (val >= 0.f) atomicMax((int*)addr, __float_as_int(val));
  else            atomicMin((unsigned int*)addr, (unsigned int)__float_as_int(val));
}

__global__ void k_fill_neg_inf(float* p, int n) {
  int i = blockIdx.x * blockDim.x + threadIdx.x;
  if (i < n) p[i] = __int_as_float(0xFF800000);
}

__global__ void k_fix_m(float* p, int n) {
  int i = blockIdx.x * blockDim.x + threadIdx.x;
  if (i < n) { float v = p[i]; if (!(v >= -3.0e38f)) p[i] = 0.f; }
}

__global__ void k_node_dots(const float* __restrict__ hw, int C,
                            const float* __restrict__ as, const float* __restrict__ ad,
                            int n, float* __restrict__ als, float* __restrict__ ald) {
  int i = blockIdx.x * blockDim.x + threadIdx.x;
  if (i >= n) return;
  const float* row = hw + (size_t)i * C;
  float s = 0.f, d = 0.f;
  for (int c = 0; c < C; ++c) { float h = row[c]; s += h * as[c]; d += h * ad[c]; }
  als[i] = s; ald[i] = d;
}

__global__ void k_edge_logit(const int* __restrict__ esrc, const int* __restrict__ edst,
                             int E, int Etot, const float* __restrict__ als,
                             const float* __restrict__ ald, float* __restrict__ elog,
                             float* __restrict__ m) {
  int e = blockIdx.x * blockDim.x + threadIdx.x;
  if (e >= Etot) return;
  int s = (e < E) ? esrc[e] : (e - E);
  int d = (e < E) ? edst[e] : (e - E);
  float x = als[s] + ald[d];
  x = (x > 0.f) ? x : 0.2f * x;                 // leaky_relu 0.2
  elog[e] = x;
  atomicMaxFloat(m + d, x);
}

__global__ void k_edge_exp(const int* __restrict__ edst, int E, int Etot,
                           const float* __restrict__ m, float* __restrict__ elog,
                           float* __restrict__ denom) {
  int e = blockIdx.x * blockDim.x + threadIdx.x;
  if (e >= Etot) return;
  int d = (e < E) ? edst[e] : (e - E);
  float ex = __expf(elog[e] - m[d]);
  elog[e] = ex;
  atomicAdd(denom + d, ex);
}

__global__ void k_edge_agg(const int* __restrict__ esrc, const int* __restrict__ edst, int E,
                           const float* __restrict__ ex, const float* __restrict__ denom,
                           const float* __restrict__ hw, int C, float* __restrict__ agg) {
  int e = blockIdx.x;                            // one block per edge, blockDim == C
  int s = (e < E) ? esrc[e] : (e - E);
  int d = (e < E) ? edst[e] : (e - E);
  float alpha = ex[e] / (denom[d] + 1e-16f);
  int c = threadIdx.x;
  atomicAdd(agg + (size_t)d * C + c, alpha * hw[(size_t)s * C + c]);
}

__global__ void k_gat_fin(const float* __restrict__ agg, const float* __restrict__ bias,
                          int n, int C, half_t* __restrict__ out, int ldo, int col0) {
  long idx = (long)blockIdx.x * blockDim.x + threadIdx.x;
  if (idx >= (long)n * C) return;
  int r = (int)(idx / C), c = (int)(idx - (long)r * C);
  float v = fmaxf(agg[idx] + bias[c], 0.f);
  out[(size_t)r * ldo + col0 + c] = (half_t)v;
}

// ---------------- LSTM-gate elementwise ----------------
__global__ void k_bsum(const float* a, const float* b, float* out) {
  int j = threadIdx.x; out[j] = a[j] + b[j];
}

__global__ void k_lstm(const float* __restrict__ gates, int n,
                       half_t* __restrict__ out, int ldo, int col0) {
  long idx = (long)blockIdx.x * blockDim.x + threadIdx.x;
  if (idx >= (long)n * 64) return;
  int r = (int)(idx / 64), j = (int)(idx - (long)r * 64);
  const float* g = gates + (size_t)r * 256;
  float c = sigmoidf_(g[j]) * tanhf(g[128 + j]);        // i,g  (f gate unused)
  float h = sigmoidf_(g[192 + j]) * tanhf(c);           // o
  out[(size_t)r * ldo + col0 + j] = (half_t)h;
}

// ---------------- host side ----------------
static inline void gemm_h16(hipStream_t st, int act, const half_t* A, const half_t* W,
                            const float* bias, half_t* C, int ldc, int col0,
                            int M, int Kpad, int NoutPad, int NoutStore) {
  dim3 grid(((M + 15) / 16 + 7) / 8, NoutPad / 16), block(256);
  if (act == ACT_SIG)
    k_gemm_wmma<ACT_SIG, half_t><<<grid, block, 0, st>>>(A, W, bias, C, ldc, col0, M, Kpad, NoutStore);
  else
    k_gemm_wmma<ACT_RELU, half_t><<<grid, block, 0, st>>>(A, W, bias, C, ldc, col0, M, Kpad, NoutStore);
}

static inline void gemm_f32(hipStream_t st, const half_t* A, const half_t* W,
                            const float* bias, float* C, int ldc, int col0,
                            int M, int Kpad, int NoutPad, int NoutStore) {
  dim3 grid(((M + 15) / 16 + 7) / 8, NoutPad / 16), block(256);
  k_gemm_wmma<ACT_NONE, float><<<grid, block, 0, st>>>(A, W, bias, C, ldc, col0, M, Kpad, NoutStore);
}

static void gat_layer(hipStream_t st, const half_t* Xin, int Kpad, const half_t* Wh,
                      const float* a_s, const float* a_d, const float* bias, int C,
                      half_t* Xout, int ldo, int col0,
                      float* hw, float* als, float* ald, float* mseg, float* den,
                      float* elog, float* agg, const int* esrc, const int* edst) {
  const int N = NNODES, E = NEDGES, Etot = E + N;
  gemm_f32(st, Xin, Wh, nullptr, hw, C, 0, N, Kpad, C, C);
  k_node_dots<<<(N + 255) / 256, 256, 0, st>>>(hw, C, a_s, a_d, N, als, ald);
  k_fill_neg_inf<<<(N + 255) / 256, 256, 0, st>>>(mseg, N);
  hipMemsetAsync(den, 0, (size_t)N * 4, st);
  hipMemsetAsync(agg, 0, (size_t)N * C * 4, st);
  k_edge_logit<<<(Etot + 255) / 256, 256, 0, st>>>(esrc, edst, E, Etot, als, ald, elog, mseg);
  k_fix_m<<<(N + 255) / 256, 256, 0, st>>>(mseg, N);
  k_edge_exp<<<(Etot + 255) / 256, 256, 0, st>>>(edst, E, Etot, mseg, elog, den);
  k_edge_agg<<<Etot, C, 0, st>>>(esrc, edst, E, elog, den, hw, C, agg);
  k_gat_fin<<<(int)(((long)N * C + 255) / 256), 256, 0, st>>>(agg, bias, N, C, Xout, ldo, col0);
}

extern "C" void kernel_launch(void* const* d_in, const int* in_sizes, int n_in,
                              void* d_out, int out_size, void* d_ws, size_t ws_size,
                              hipStream_t stream) {
  (void)in_sizes; (void)n_in; (void)out_size; (void)ws_size;
  const int N = NNODES, E = NEDGES;

  const float* x_all   = (const float*)d_in[0];
  const int*   eidx    = (const int*)d_in[1];
  const int*   esrc    = eidx;
  const int*   edst    = eidx + E;
  const float* sep1_w  = (const float*)d_in[2],  *sep1_b = (const float*)d_in[3];
  const float* sep2_w  = (const float*)d_in[4],  *sep2_b = (const float*)d_in[5];
  const float* sep3_w  = (const float*)d_in[6],  *sep3_b = (const float*)d_in[7];
  const float* sep4_w  = (const float*)d_in[8],  *sep4_b = (const float*)d_in[9];
  const float* sep5_w  = (const float*)d_in[10], *sep5_b = (const float*)d_in[11];
  const float* sep6_w  = (const float*)d_in[12], *sep6_b = (const float*)d_in[13];
  const float* sep7_w  = (const float*)d_in[14], *sep7_b = (const float*)d_in[15];
  const float* sep8_w  = (const float*)d_in[16], *sep8_b = (const float*)d_in[17];
  const float* conv1_w = (const float*)d_in[18], *conv1_as = (const float*)d_in[19];
  const float* conv1_ad= (const float*)d_in[20], *conv1_b  = (const float*)d_in[21];
  const float* conv2_w = (const float*)d_in[22], *conv2_as = (const float*)d_in[23];
  const float* conv2_ad= (const float*)d_in[24], *conv2_b  = (const float*)d_in[25];
  const float* conv3_w = (const float*)d_in[26], *conv3_as = (const float*)d_in[27];
  const float* conv3_ad= (const float*)d_in[28], *conv3_b  = (const float*)d_in[29];
  const float* lstm_wih= (const float*)d_in[30];
  const float* lstm_bih= (const float*)d_in[32], *lstm_bhh = (const float*)d_in[33];
  const float* lin1_w  = (const float*)d_in[34], *lin1_b = (const float*)d_in[35];
  const float* lin2_w  = (const float*)d_in[36], *lin2_b = (const float*)d_in[37];
  const float* lin3_w  = (const float*)d_in[38], *lin3_b = (const float*)d_in[39];

  // ---- workspace bump allocator ----
  uintptr_t base = (uintptr_t)d_ws;
  auto alloc = [&](size_t bytes) -> void* {
    uintptr_t p = base; base += (bytes + 255) & ~(size_t)255; return (void*)p;
  };
  half_t* X    = (half_t*)alloc((size_t)N * 384 * 2);   // sep concat (f16, Kpad=384)
  half_t* XC   = (half_t*)alloc((size_t)N * 128 * 2);   // [gat3 | lstm_h]
  half_t* X2   = (half_t*)alloc((size_t)N * 128 * 2);
  half_t* X3   = (half_t*)alloc((size_t)N * 128 * 2);
  float*  hw   = (float*) alloc((size_t)N * 128 * 4);
  float*  agg  = (float*) alloc((size_t)N * 128 * 4);
  float*  als  = (float*) alloc((size_t)N * 4);
  float*  ald  = (float*) alloc((size_t)N * 4);
  float*  mseg = (float*) alloc((size_t)N * 4);
  float*  den  = (float*) alloc((size_t)N * 4);
  float*  elog = (float*) alloc((size_t)(E + N) * 4);
  half_t* Y1   = (half_t*)alloc((size_t)N * 64 * 2);
  half_t* Y2   = (half_t*)alloc((size_t)N * 64 * 2);
  half_t* At   = (half_t*)alloc((size_t)N * 32 * 2);    // lstm input, Kpad=32
  half_t* w1h  = (half_t*)alloc(64 * 32 * 2);
  half_t* w2h  = (half_t*)alloc(64 * 32 * 2);
  half_t* w3h  = (half_t*)alloc(128 * 64 * 2);
  half_t* w4h  = (half_t*)alloc(128 * 384 * 2);
  half_t* w5h  = (half_t*)alloc(64 * 64 * 2);
  half_t* w6h  = (half_t*)alloc(64 * 64 * 2);
  half_t* w7h  = (half_t*)alloc(128 * 128 * 2);
  half_t* w8h  = (half_t*)alloc(128 * 128 * 2);
  half_t* c1h  = (half_t*)alloc(128 * 384 * 2);
  half_t* c2h  = (half_t*)alloc(128 * 128 * 2);
  half_t* c3h  = (half_t*)alloc(64 * 128 * 2);
  half_t* lwh  = (half_t*)alloc(256 * 32 * 2);
  half_t* l1h  = (half_t*)alloc(64 * 128 * 2);
  half_t* l2h  = (half_t*)alloc(64 * 64 * 2);
  half_t* l3h  = (half_t*)alloc(64 * 64 * 2);           // lin3 padded 56->64 rows
  float*  bsum = (float*) alloc(256 * 4);
  // scratch region S: sep-stage A/H buffers; later reused for LSTM gates
  char* S = (char*)alloc((size_t)90 * 1000 * 1000);
  size_t so = 0;
  auto salloc = [&](size_t b) -> void* { void* p = S + so; so += (b + 255) & ~(size_t)255; return p; };
  half_t* A1 = (half_t*)salloc((size_t)N * 32 * 2);
  half_t* A2 = (half_t*)salloc((size_t)N * 32 * 2);
  half_t* A3 = (half_t*)salloc((size_t)N * 64 * 2);
  half_t* A4 = (half_t*)salloc((size_t)N * 384 * 2);
  half_t* H1 = (half_t*)salloc((size_t)N * 64 * 2);
  half_t* H2 = (half_t*)salloc((size_t)N * 64 * 2);
  half_t* H3 = (half_t*)salloc((size_t)N * 128 * 2);
  half_t* H4 = (half_t*)salloc((size_t)N * 128 * 2);
  float*  gates = (float*)S;                            // aliases A1..A4 (dead by then)

  // ---- phase 0: f16 conversions (zero-padded K) ----
  auto cvt_slice = [&](int c0, int w, half_t* dst, int kpad) {
    long n = (long)N * kpad;
    k_cvt_slice<<<(int)((n + 255) / 256), 256, 0, stream>>>(x_all, 441, c0, w, dst, kpad, N);
  };
  cvt_slice(1, 2, A1, 32);
  cvt_slice(3, 13, A2, 32);
  cvt_slice(16, 40, A3, 64);
  cvt_slice(56, 365, A4, 384);
  cvt_slice(421, 20, At, 32);
  struct WC { const float* s; int no, ki; half_t* d; int nop, kp; };
  const WC wcs[] = {
    {sep1_w,64,2,w1h,64,32},   {sep2_w,64,13,w2h,64,32},   {sep3_w,128,40,w3h,128,64},
    {sep4_w,128,365,w4h,128,384},{sep5_w,64,64,w5h,64,64}, {sep6_w,64,64,w6h,64,64},
    {sep7_w,128,128,w7h,128,128},{sep8_w,128,128,w8h,128,128},
    {conv1_w,128,384,c1h,128,384},{conv2_w,128,128,c2h,128,128},{conv3_w,64,128,c3h,64,128},
    {lstm_wih,256,20,lwh,256,32},{lin1_w,64,128,l1h,64,128},{lin2_w,64,64,l2h,64,64},
    {lin3_w,56,64,l3h,64,64},
  };
  for (const WC& c : wcs)
    k_cvt_w<<<(c.nop * c.kp + 255) / 256, 256, 0, stream>>>(c.s, c.no, c.ki, c.d, c.nop, c.kp);

  // ---- phase 1: sep MLPs -> X (N,384) ----
  gemm_h16(stream, ACT_SIG,  A1, w1h, sep1_b, H1, 64, 0,   N, 32,  64, 64);
  gemm_h16(stream, ACT_RELU, H1, w5h, sep5_b, X, 384, 0,   N, 64,  64, 64);
  gemm_h16(stream, ACT_SIG,  A2, w2h, sep2_b, H2, 64, 0,   N, 32,  64, 64);
  gemm_h16(stream, ACT_RELU, H2, w6h, sep6_b, X, 384, 64,  N, 64,  64, 64);
  gemm_h16(stream, ACT_SIG,  A3, w3h, sep3_b, H3, 128, 0,  N, 64,  128, 128);
  gemm_h16(stream, ACT_RELU, H3, w7h, sep7_b, X, 384, 128, N, 128, 128, 128);
  gemm_h16(stream, ACT_SIG,  A4, w4h, sep4_b, H4, 128, 0,  N, 384, 128, 128);
  gemm_h16(stream, ACT_RELU, H4, w8h, sep8_b, X, 384, 256, N, 128, 128, 128);

  // ---- phases 2-4: GAT layers ----
  gat_layer(stream, X,  384, c1h, conv1_as, conv1_ad, conv1_b, 128, X2, 128, 0,
            hw, als, ald, mseg, den, elog, agg, esrc, edst);
  gat_layer(stream, X2, 128, c2h, conv2_as, conv2_ad, conv2_b, 128, X3, 128, 0,
            hw, als, ald, mseg, den, elog, agg, esrc, edst);
  gat_layer(stream, X3, 128, c3h, conv3_as, conv3_ad, conv3_b, 64,  XC, 128, 0,
            hw, als, ald, mseg, den, elog, agg, esrc, edst);

  // ---- phase 5: LSTM gates -> XC[:,64:128] ----
  k_bsum<<<1, 256, 0, stream>>>(lstm_bih, lstm_bhh, bsum);
  gemm_f32(stream, At, lwh, bsum, gates, 256, 0, N, 32, 256, 256);
  k_lstm<<<(int)(((long)N * 64 + 255) / 256), 256, 0, stream>>>(gates, N, XC, 128, 64);

  // ---- phase 6: output head ----
  gemm_h16(stream, ACT_RELU, XC, l1h, lin1_b, Y1, 64, 0, N, 128, 64, 64);
  gemm_h16(stream, ACT_RELU, Y1, l2h, lin2_b, Y2, 64, 0, N, 64, 64, 64);
  gemm_f32(stream, Y2, l3h, lin3_b, (float*)d_out, 56, 0, N, 64, 64, 56);
}
